// TransformerBlock_39728447488203
// MI455X (gfx1250) — matrix operand order repaired and checked by reading, not compiled, not run on hardware
//
#include <hip/hip_runtime.h>
#include <hip/hip_bf16.h>

typedef _Float16 half_t;
typedef __attribute__((ext_vector_type(16))) _Float16 v16h;
typedef __attribute__((ext_vector_type(8)))  _Float16 v8h;
typedef __attribute__((ext_vector_type(8)))  float    v8f;

#define E_ 1024
#define T_ 2048
#define H_ 16
#define HD_ 64
#define FF_ 4096
#define B_ 4
#define ROWS (B_ * T_)   // 8192

// ---------------------------------------------------------------------------
// A-operand fragment for v_wmma_f32_16x16x32_f16 (ISA 7.12.2, 16-bit A 16x32):
// lane holds row m = lane&15; lanes 0-15 carry K = {0..7, 16..23},
// lanes 16-31 carry K = {8..15, 24..31} (two 16B chunks).
// `p` points at element [row][k0] of a row-major (K-contiguous) matrix.
// ---------------------------------------------------------------------------
__device__ inline v16h frag_a(const half_t* p, int g) {
    union { v16h v; v8h h[2]; } u;
    u.h[0] = *(const v8h*)(p + g * 8);
    u.h[1] = *(const v8h*)(p + 16 + g * 8);
    return u.v;
}

// ---------------------------------------------------------------------------
// B-operand fragment (same K order as frag_a): lane holds col n = lane&15;
// lanes 0-15 carry K = {0..7, 16..23}, lanes 16-31 carry K = {8..15, 24..31}
// (two 16B chunks). `p` points at element [n][k0] of the transposed (N-major,
// K-contiguous) B matrix.
// ---------------------------------------------------------------------------
__device__ inline v16h frag_b(const half_t* p, int g) {
    union { v16h v; v8h h[2]; } u;
    u.h[0] = *(const v8h*)(p + g * 8);
    u.h[1] = *(const v8h*)(p + 16 + g * 8);
    return u.v;
}

__device__ inline v8f wmma_f16(v16h a, v16h b, v8f c) {
    return __builtin_amdgcn_wmma_f32_16x16x32_f16(false, a, false, b, (short)0, c,
                                                  false, false);
}

// ---------------------------------------------------------------------------
// LayerNorm (row = 1024 floats) -> f16. One block per row, 256 threads x 4.
// ---------------------------------------------------------------------------
__global__ __launch_bounds__(256) void ln_kernel(const float* __restrict__ x,
                                                 const float* __restrict__ gam,
                                                 const float* __restrict__ bet,
                                                 half_t* __restrict__ out) {
    __shared__ float red[16];
    const int row = blockIdx.x;
    const int tid = threadIdx.x;
    const float4 v = ((const float4*)(x + (size_t)row * E_))[tid];
    float s  = v.x + v.y + v.z + v.w;
    float sq = v.x * v.x + v.y * v.y + v.z * v.z + v.w * v.w;
#pragma unroll
    for (int off = 16; off; off >>= 1) {
        s  += __shfl_xor(s,  off, 32);
        sq += __shfl_xor(sq, off, 32);
    }
    if ((tid & 31) == 0) { red[tid >> 5] = s; red[8 + (tid >> 5)] = sq; }
    __syncthreads();
    float S = 0.f, SQ = 0.f;
#pragma unroll
    for (int i = 0; i < 8; i++) { S += red[i]; SQ += red[8 + i]; }
    const float mean = S * (1.0f / E_);
    const float var  = SQ * (1.0f / E_) - mean * mean;
    const float rstd = rsqrtf(var + 1e-5f);
    const int i0 = tid * 4;
    half_t* o = out + (size_t)row * E_ + i0;
    o[0] = (half_t)((v.x - mean) * rstd * gam[i0 + 0] + bet[i0 + 0]);
    o[1] = (half_t)((v.y - mean) * rstd * gam[i0 + 1] + bet[i0 + 1]);
    o[2] = (half_t)((v.z - mean) * rstd * gam[i0 + 2] + bet[i0 + 2]);
    o[3] = (half_t)((v.w - mean) * rstd * gam[i0 + 3] + bet[i0 + 3]);
}

// ---------------------------------------------------------------------------
// Weight conversion: [R rows(in), C cols(out)] fp32 -> transposed f16 [C][R]
// ---------------------------------------------------------------------------
__global__ __launch_bounds__(256) void wt_conv(const float* __restrict__ w,
                                               half_t* __restrict__ out,
                                               int R, int C) {
    const size_t i = (size_t)blockIdx.x * 256 + threadIdx.x;
    const int r = (int)(i % R);
    const int n = (int)(i / R);
    out[i] = (half_t)w[(size_t)r * C + n];
}

// wq/wk/wv [H][E][HD] fp32 -> packed Bt f16 [3*1024][1024]; n = type*1024+h*64+d
__global__ __launch_bounds__(256) void wqkv_conv(const float* __restrict__ wq,
                                                 const float* __restrict__ wk,
                                                 const float* __restrict__ wv,
                                                 half_t* __restrict__ out) {
    const size_t i = (size_t)blockIdx.x * 256 + threadIdx.x;  // 3*1024*1024
    const int c    = (int)(i & 1023);
    const int n    = (int)(i >> 10);
    const int type = n >> 10;
    const int nn   = n & 1023;
    const int head = nn >> 6;
    const int d    = nn & 63;
    const float* w = (type == 0) ? wq : (type == 1) ? wk : wv;
    out[i] = (half_t)w[((size_t)head * E_ + c) * HD_ + d];
}

// ---------------------------------------------------------------------------
// WMMA GEMM: O[M,N] = A[M,K] * Bt[N,K]^T, block tile 128x128, wave tile 64x32.
// MODE 0: QKV scatter epilogue (q scaled, V stored transposed [HD][T])
// MODE 1: f16 out, + bias + ReLU                     (FF1)
// MODE 2: f32 out = res + acc + bias                 (proj / FF2)
// ---------------------------------------------------------------------------
template <int MODE>
__global__ __launch_bounds__(256) void gemm_kernel(
    const half_t* __restrict__ A, int lda, const half_t* __restrict__ Bt, int ldb,
    const float* __restrict__ bias, const float* __restrict__ res,
    float* __restrict__ outF, half_t* __restrict__ outH,
    half_t* __restrict__ qo, half_t* __restrict__ ko, half_t* __restrict__ vo,
    int M, int N, int K, float qscale) {
    const int lane = threadIdx.x & 31;
    const int w    = threadIdx.x >> 5;
    const int g    = lane >> 4;
    const int mi   = lane & 15;
    const int m0   = blockIdx.y * 128 + (w >> 2) * 64;
    const int n0   = blockIdx.x * 128 + (w & 3) * 32;

    v8f acc[4][2] = {};
    for (int k0 = 0; k0 < K; k0 += 32) {
        v16h a[4], b[2];
#pragma unroll
        for (int i = 0; i < 4; i++)
            a[i] = frag_a(A + (size_t)(m0 + i * 16 + mi) * lda + k0, g);
#pragma unroll
        for (int j = 0; j < 2; j++)
            b[j] = frag_b(Bt + (size_t)(n0 + j * 16 + mi) * ldb + k0, g);
#pragma unroll
        for (int i = 0; i < 4; i++)
#pragma unroll
            for (int j = 0; j < 2; j++) acc[i][j] = wmma_f16(a[i], b[j], acc[i][j]);
    }

#pragma unroll
    for (int i = 0; i < 4; i++)
#pragma unroll
        for (int j = 0; j < 2; j++)
#pragma unroll
            for (int r = 0; r < 8; r++) {
                const int m = m0 + i * 16 + r + 8 * g;
                const int n = n0 + j * 16 + mi;
                float val   = acc[i][j][r];
                if (MODE == 0) {
                    const int type = n >> 10, nn = n & 1023;
                    const int head = nn >> 6, d = nn & 63;
                    const int bb = m >> 11, t = m & 2047;
                    const size_t bh = (size_t)(bb * H_ + head);
                    if (type == 0)
                        qo[(bh * T_ + t) * HD_ + d] = (half_t)(val * qscale);
                    else if (type == 1)
                        ko[(bh * T_ + t) * HD_ + d] = (half_t)val;
                    else
                        vo[(bh * HD_ + d) * T_ + t] = (half_t)val;
                } else if (MODE == 1) {
                    val += bias[n];
                    outH[(size_t)m * N + n] = (half_t)fmaxf(val, 0.0f);
                } else {
                    outF[(size_t)m * N + n] = res[(size_t)m * N + n] + val + bias[n];
                }
            }
}

// ---------------------------------------------------------------------------
// Flash attention per (b,h). Block = 128 queries, each wave owns 16 rows.
// scores = Q'K^T (scale pre-folded into Q); online softmax; P staged via LDS
// to convert C-layout -> A-layout for the PV WMMA.
// ---------------------------------------------------------------------------
__global__ __launch_bounds__(256) void attn_kernel(const half_t* __restrict__ q,
                                                   const half_t* __restrict__ k,
                                                   const half_t* __restrict__ vT,
                                                   half_t* __restrict__ attn) {
    __shared__ half_t Pbuf[8 * 16 * 64];
    const int lane = threadIdx.x & 31;
    const int wid  = threadIdx.x >> 5;
    const int g    = lane >> 4;
    const int mi   = lane & 15;
    const int bh   = blockIdx.y;
    const int b    = bh >> 4;
    const int h    = bh & 15;
    const int qm   = blockIdx.x * 128 + wid * 16;

    const half_t* Q  = q  + (size_t)bh * T_ * HD_;
    const half_t* Kp = k  + (size_t)bh * T_ * HD_;
    const half_t* Vt = vT + (size_t)bh * HD_ * T_;
    half_t* Pw = Pbuf + wid * 16 * 64;

    v16h aQ[2];
    aQ[0] = frag_a(Q + (size_t)(qm + mi) * HD_ + 0,  g);
    aQ[1] = frag_a(Q + (size_t)(qm + mi) * HD_ + 32, g);

    v8f o[4] = {};
    float mrun[8], lrun[8];
#pragma unroll
    for (int r = 0; r < 8; r++) { mrun[r] = -1e30f; lrun[r] = 0.f; }

    for (int s0 = 0; s0 < T_; s0 += 64) {
        v8f s[4] = {};
#pragma unroll
        for (int j = 0; j < 4; j++) {
            const half_t* kr = Kp + (size_t)(s0 + j * 16 + mi) * HD_;  // B-op: n=key
            s[j] = wmma_f16(aQ[0], frag_b(kr + 0,  g), s[j]);
            s[j] = wmma_f16(aQ[1], frag_b(kr + 32, g), s[j]);
        }
#pragma unroll
        for (int r = 0; r < 8; r++) {
            float mx = s[0][r];
#pragma unroll
            for (int j = 1; j < 4; j++) mx = fmaxf(mx, s[j][r]);
#pragma unroll
            for (int off = 8; off; off >>= 1) mx = fmaxf(mx, __shfl_xor(mx, off, 32));
            const float mnew = fmaxf(mrun[r], mx);
            const float corr = __expf(mrun[r] - mnew);
            mrun[r] = mnew;
            float rs = 0.f;
#pragma unroll
            for (int j = 0; j < 4; j++) {
                const float p = __expf(s[j][r] - mnew);
                rs += p;
                Pw[(r + 8 * g) * 64 + j * 16 + mi] = (half_t)p;
            }
#pragma unroll
            for (int off = 8; off; off >>= 1) rs += __shfl_xor(rs, off, 32);
            lrun[r] = lrun[r] * corr + rs;
#pragma unroll
            for (int j = 0; j < 4; j++) o[j][r] *= corr;
        }
        // Same-wave DS ops are in-order; wait for stores before cross-lane reads.
        asm volatile("s_wait_dscnt 0" ::: "memory");
        v16h pa0 = frag_a(Pw + (size_t)mi * 64 + 0,  g);
        v16h pa1 = frag_a(Pw + (size_t)mi * 64 + 32, g);
#pragma unroll
        for (int j = 0; j < 4; j++) {
            const half_t* vr = Vt + (size_t)(j * 16 + mi) * T_ + s0;  // B-op: n=d
            o[j] = wmma_f16(pa0, frag_b(vr + 0,  g), o[j]);
            o[j] = wmma_f16(pa1, frag_b(vr + 32, g), o[j]);
        }
    }
#pragma unroll
    for (int j = 0; j < 4; j++)
#pragma unroll
        for (int r = 0; r < 8; r++) {
            const int d = j * 16 + mi;
            const int t = qm + r + 8 * g;
            const float val = o[j][r] / lrun[r];
            attn[((size_t)(b * T_ + t)) * E_ + h * HD_ + d] = (half_t)val;
        }
}

// ---------------------------------------------------------------------------
extern "C" void kernel_launch(void* const* d_in, const int* in_sizes, int n_in,
                              void* d_out, int out_size, void* d_ws, size_t ws_size,
                              hipStream_t stream) {
    const float* x      = (const float*)d_in[0];
    const float* wq     = (const float*)d_in[1];
    const float* wk     = (const float*)d_in[2];
    const float* wv     = (const float*)d_in[3];
    const float* w_proj = (const float*)d_in[4];
    const float* b_proj = (const float*)d_in[5];
    const float* w1     = (const float*)d_in[6];
    const float* b1     = (const float*)d_in[7];
    const float* w2     = (const float*)d_in[8];
    const float* b2     = (const float*)d_in[9];
    const float* g1     = (const float*)d_in[10];
    const float* bt1    = (const float*)d_in[11];
    const float* g2     = (const float*)d_in[12];
    const float* bt2    = (const float*)d_in[13];

    char* base = (char*)d_ws;
    float*  x1    = (float*)base;                                   // 32 MiB
    half_t* hbuf  = (half_t*)(base + (size_t)33554432);             // 16 MiB (h / h2)
    half_t* wbuf  = (half_t*)(base + (size_t)33554432 + 16777216);  //  8 MiB (weights)
    char*   big   = base + (size_t)33554432 + 16777216 + 8388608;   // 64 MiB
    half_t* qbuf  = (half_t*)big;
    half_t* kbuf  = (half_t*)(big + (size_t)16777216);
    half_t* vbuf  = (half_t*)(big + (size_t)2 * 16777216);
    half_t* abuf  = (half_t*)(big + (size_t)3 * 16777216);
    half_t* ffmid = (half_t*)big;  // overlays q/k/vT/attn after they are dead

    const float qscale = 0.03125f;  // 1024^-0.5

    // 1) h = layernorm(x) -> f16
    ln_kernel<<<ROWS, 256, 0, stream>>>(x, g1, bt1, hbuf);
    // 2) pack wq/wk/wv -> Bt [3072][1024]
    wqkv_conv<<<(3 * E_ * E_) / 256, 256, 0, stream>>>(wq, wk, wv, wbuf);
    // 3) QKV GEMM [8192,1024] x [1024,3072] with scatter epilogue
    gemm_kernel<0><<<dim3(3 * E_ / 128, ROWS / 128), 256, 0, stream>>>(
        hbuf, E_, wbuf, E_, nullptr, nullptr, nullptr, nullptr,
        qbuf, kbuf, vbuf, ROWS, 3 * E_, E_, qscale);
    // 4) attention -> abuf f16 [ROWS][E]
    attn_kernel<<<dim3(T_ / 128, B_ * H_), 256, 0, stream>>>(qbuf, kbuf, vbuf, abuf);
    // 5) w_proj -> Bt [1024][1024]
    wt_conv<<<(E_ * E_) / 256, 256, 0, stream>>>(w_proj, wbuf, E_, E_);
    // 6) x1 = x + attn @ w_proj + b_proj
    gemm_kernel<2><<<dim3(E_ / 128, ROWS / 128), 256, 0, stream>>>(
        abuf, E_, wbuf, E_, b_proj, x, x1, nullptr, nullptr, nullptr, nullptr,
        ROWS, E_, E_, 0.f);
    // 7) h2 = layernorm(x1) -> f16 (reuse hbuf)
    ln_kernel<<<ROWS, 256, 0, stream>>>(x1, g2, bt2, hbuf);
    // 8) w1 -> Bt [4096][1024]
    wt_conv<<<(E_ * FF_) / 256, 256, 0, stream>>>(w1, wbuf, E_, FF_);
    // 9) ffmid = relu(h2 @ w1 + b1) -> f16 [8192][4096]
    gemm_kernel<1><<<dim3(FF_ / 128, ROWS / 128), 256, 0, stream>>>(
        hbuf, E_, wbuf, E_, b1, nullptr, nullptr, ffmid, nullptr, nullptr, nullptr,
        ROWS, FF_, E_, 0.f);
    // 10) w2 -> Bt [1024][4096]
    wt_conv<<<(FF_ * E_) / 256, 256, 0, stream>>>(w2, wbuf, FF_, E_);
    // 11) out = x1 + ffmid @ w2 + b2  (fp32, straight to d_out)
    gemm_kernel<2><<<dim3(E_ / 128, ROWS / 128), 256, 0, stream>>>(
        ffmid, FF_, wbuf, FF_, b2, x1, (float*)d_out, nullptr, nullptr, nullptr,
        nullptr, ROWS, E_, FF_, 0.f);
    (void)in_sizes; (void)n_in; (void)out_size; (void)ws_size;
}
